// EnhancedTransformerLayer_26912265076926
// MI455X (gfx1250) — matrix operand order repaired and checked by reading, not compiled, not run on hardware
//
#include <hip/hip_runtime.h>

typedef _Float16 half_t;
typedef __attribute__((ext_vector_type(16))) _Float16 v16h;
typedef __attribute__((ext_vector_type(8)))  _Float16 v8h;
typedef __attribute__((ext_vector_type(4)))  _Float16 v4h;
typedef __attribute__((ext_vector_type(8)))  float    v8f;

#define B_   2
#define S_   2048
#define E_   1024
#define H_   16
#define D_   64
#define NE_  8
#define MTOK (B_ * S_)

union Frag16 { v16h v; v8h h[2]; };

__device__ __forceinline__ v8f wmma_f16(v16h a, v16h b, v8f c) {
  // D = A(16x32 f16) * B(32x16 f16) + C(16x16 f32)
  return __builtin_amdgcn_wmma_f32_16x16x32_f16(false, a, false, b, (short)0, c,
                                                false, false);
}

// CDNA5 async global->LDS copy (16B per lane), tracked by ASYNCcnt.
// dsaddr = wave LDS base + per-lane VGPR offset; GV addressing (64-bit VA).
__device__ __forceinline__ void async_load_b128(unsigned lds_off,
                                                const half_t* gaddr) {
  asm volatile("global_load_async_to_lds_b128 %0, %1, off" ::"v"(lds_off),
               "v"(gaddr)
               : "memory");
}
__device__ __forceinline__ void async_wait0() {
#if __has_builtin(__builtin_amdgcn_s_wait_asynccnt)
  __builtin_amdgcn_s_wait_asynccnt(0);
#else
  asm volatile("s_wait_asynccnt 0" ::: "memory");
#endif
}
__device__ __forceinline__ unsigned lds_off_u32(const void* p) {
  return (unsigned)(size_t)p;  // low 32 bits of generic shared ptr = LDS offset
}

// ---------------------------------------------------------------------------
// fp32 -> fp16 bulk cast (weights).  4 elements / thread.
// ---------------------------------------------------------------------------
__global__ __launch_bounds__(256) void cast_f16_kernel(
    const float* __restrict__ s, half_t* __restrict__ d) {
  int i = (blockIdx.x * 256 + threadIdx.x) * 4;
  float4 f = *(const float4*)(s + i);
  v4h o;
  o[0] = (half_t)f.x; o[1] = (half_t)f.y;
  o[2] = (half_t)f.z; o[3] = (half_t)f.w;
  *(v4h*)(d + i) = o;
}

// ---------------------------------------------------------------------------
// RoPE + fp32->fp16 cast.  One thread per rotation pair (b,s,h,i).
// ---------------------------------------------------------------------------
__global__ __launch_bounds__(256) void rope_cast_kernel(
    const float* __restrict__ x, half_t* __restrict__ x_h,
    half_t* __restrict__ qin_h) {
  int idx = blockIdx.x * 256 + threadIdx.x;
  int tok = idx >> 9;
  int rem = idx & 511;
  int h   = rem >> 5;
  int i   = rem & 31;
  int pos = tok & (S_ - 1);
  size_t base = (size_t)tok * E_ + h * D_ + 2 * i;
  float x1 = x[base], x2 = x[base + 1];
  float inv = __powf(10000.f, -((float)(2 * i) / (float)D_));
  float ang = (float)pos * inv;
  float sn, cs;
  __sincosf(ang, &sn, &cs);
  qin_h[base]     = (half_t)(x1 * cs - x2 * sn);
  qin_h[base + 1] = (half_t)(x1 * sn + x2 * cs);
  x_h[base]     = (half_t)x1;
  x_h[base + 1] = (half_t)x2;
}

// ---------------------------------------------------------------------------
// Tiled WMMA GEMM:  C[M,N] = A[M,K](f16) * W[N,K](f16)^T + bias[N]
// 256 threads = 8 waves (4x2), block tile 128x128, wave tile 32x64, BK=64.
// Both operand tiles streamed with global_load_async_to_lds_b128.
// MODE 0: write f16    MODE 1: write f32
// MODE 2: MoE -> outf[m,n] (+)= gsel[m*8+e] * (acc + bias)  (INIT selects =/+=)
// ---------------------------------------------------------------------------
template <int MODE, bool INIT>
__global__ __launch_bounds__(256, 2) void gemm_wmma_kernel(
    const half_t* __restrict__ A, const half_t* __restrict__ W,
    const float* __restrict__ bias, half_t* __restrict__ outh,
    float* __restrict__ outf, const float* __restrict__ gsel, int eidx,
    int Ndim, int Kdim) {
  __shared__ __align__(16) half_t As[128][72];
  __shared__ __align__(16) half_t Bs[128][72];
  int tid  = threadIdx.x;
  int wave = tid >> 5, lane = tid & 31;
  int wm = wave & 3, wn = wave >> 2;   // 4 x 2 wave grid
  int lm = lane & 15, hi = lane >> 4;
  int ka = hi << 3;                    // A-frag k base (ISA A 16x32 layout)
  int kb = hi << 3;                    // B-frag k base (same k order as A)
  int m0 = blockIdx.x * 128, n0 = blockIdx.y * 128;

  // staging: thread covers halfs [scol, scol+32) of row srow (64B = 4 x b128)
  int srow = tid >> 1;
  int scol = (tid & 1) << 5;
  const half_t* gA = A + (size_t)(m0 + srow) * Kdim + scol;
  const half_t* gW = W + (size_t)(n0 + srow) * Kdim + scol;
  unsigned ldsA = lds_off_u32(&As[srow][scol]);
  unsigned ldsB = lds_off_u32(&Bs[srow][scol]);

  v8f acc[2][4] = {};

  for (int k0 = 0; k0 < Kdim; k0 += 64) {
    __syncthreads();  // previous stage fully consumed before overwrite
#pragma unroll
    for (int c = 0; c < 4; c++) {
      async_load_b128(ldsA + c * 16, gA + k0 + c * 8);
      async_load_b128(ldsB + c * 16, gW + k0 + c * 8);
    }
    async_wait0();
    __syncthreads();

#pragma unroll
    for (int ks = 0; ks < 2; ks++) {
      int kk = ks * 32;
      Frag16 a[2], b[4];
#pragma unroll
      for (int mt = 0; mt < 2; mt++) {
        int row = wm * 32 + mt * 16 + lm;
        a[mt].h[0] = *(const v8h*)&As[row][kk + ka];
        a[mt].h[1] = *(const v8h*)&As[row][kk + ka + 16];
      }
#pragma unroll
      for (int nt = 0; nt < 4; nt++) {
        int row = wn * 64 + nt * 16 + lm;
        b[nt].h[0] = *(const v8h*)&Bs[row][kk + kb];
        b[nt].h[1] = *(const v8h*)&Bs[row][kk + kb + 16];
      }
#pragma unroll
      for (int mt = 0; mt < 2; mt++)
#pragma unroll
        for (int nt = 0; nt < 4; nt++)
          acc[mt][nt] = wmma_f16(a[mt].v, b[nt].v, acc[mt][nt]);
    }
  }

  // epilogue  (C layout: n = lm, m = r + 8*hi)
#pragma unroll
  for (int mt = 0; mt < 2; mt++) {
#pragma unroll
    for (int nt = 0; nt < 4; nt++) {
      int n = n0 + wn * 64 + nt * 16 + lm;
      float bv = bias[n];
#pragma unroll
      for (int r = 0; r < 8; r++) {
        int m = m0 + wm * 32 + mt * 16 + r + (hi << 3);
        float val = acc[mt][nt][r] + bv;
        size_t o = (size_t)m * Ndim + n;
        if (MODE == 0) {
          outh[o] = (half_t)val;
        } else if (MODE == 1) {
          outf[o] = val;
        } else {
          float g = gsel[(size_t)m * NE_ + eidx];
          if (INIT) outf[o] = g * val;
          else      outf[o] += g * val;
        }
      }
    }
  }
}

// ---------------------------------------------------------------------------
// Flash-style causal attention.  Block = 4 waves, each wave owns a 16-row
// q-tile (64 q rows / block).  K tile async-copied to LDS; V staged transposed
// (async copy cannot transpose).  Online softmax with shfl_xor reductions;
// P converted C->A layout through per-wave LDS scratch.
// ---------------------------------------------------------------------------
__global__ __launch_bounds__(128, 2) void attn_kernel(
    const half_t* __restrict__ Q, const half_t* __restrict__ K,
    const half_t* __restrict__ V, half_t* __restrict__ O) {
  __shared__ __align__(16) half_t Ks[32][72];
  __shared__ __align__(16) half_t Vt[64][40];
  __shared__ __align__(16) half_t Ps[4][16][40];

  int qblk = blockIdx.x;
  int bh   = blockIdx.y;
  int b = bh >> 4, h = bh & 15;
  int tid  = threadIdx.x;
  int wave = tid >> 5, lane = tid & 31;
  int lm = lane & 15, hi = lane >> 4;
  int ka = hi << 3, kb = hi << 3;

  size_t headoff = (size_t)b * S_ * E_ + h * D_;
  const half_t* q = Q + headoff;
  const half_t* k = K + headoff;
  const half_t* v = V + headoff;

  int q0 = qblk * 64 + wave * 16;

  Frag16 aQ[2];
#pragma unroll
  for (int c = 0; c < 2; c++) {
    const half_t* row = q + (size_t)(q0 + lm) * E_ + c * 32;
    aQ[c].h[0] = *(const v8h*)(row + ka);
    aQ[c].h[1] = *(const v8h*)(row + ka + 16);
  }

  v8f ctx[4] = {};
  float mrow[8], lrow[8];
#pragma unroll
  for (int r = 0; r < 8; r++) { mrow[r] = -1e30f; lrow[r] = 0.f; }

  int srow = tid >> 2;             // staging row 0..31
  int scol = (tid & 3) << 4;       // staging col base (halfs)
  unsigned ldsK = lds_off_u32(&Ks[srow][scol]);

  int ktmax = (qblk * 64 + 63) >> 5;
  for (int kt = 0; kt <= ktmax; kt++) {
    int kbase = kt * 32;
    __syncthreads();
    // K tile: async global->LDS (row-major)
    const half_t* krow = k + (size_t)(kbase + srow) * E_ + scol;
    async_load_b128(ldsK, krow);
    async_load_b128(ldsK + 16, krow + 8);
    // V tile: manual transposed store Vt[d][ktok]
    {
      const v8h* src = (const v8h*)(v + (size_t)(kbase + srow) * E_ + scol);
      v8h a0 = src[0], a1 = src[1];
#pragma unroll
      for (int i = 0; i < 8; i++) {
        Vt[scol + i][srow]     = a0[i];
        Vt[scol + 8 + i][srow] = a1[i];
      }
    }
    async_wait0();
    __syncthreads();

    // scores: S[16q x 32k] = Q * K^T
    v8f s[2];
#pragma unroll
    for (int j = 0; j < 2; j++) {
      int row = j * 16 + lm;
      Frag16 bK0, bK1;
      bK0.h[0] = *(const v8h*)&Ks[row][kb];
      bK0.h[1] = *(const v8h*)&Ks[row][kb + 16];
      bK1.h[0] = *(const v8h*)&Ks[row][32 + kb];
      bK1.h[1] = *(const v8h*)&Ks[row][32 + kb + 16];
      v8f z = {};
      z = wmma_f16(aQ[0].v, bK0.v, z);
      z = wmma_f16(aQ[1].v, bK1.v, z);
      s[j] = z;
    }

    const float scale = 0.125f;  // 1/sqrt(64)
#pragma unroll
    for (int j = 0; j < 2; j++)
#pragma unroll
      for (int r = 0; r < 8; r++) {
        int qi = q0 + r + (hi << 3);
        int ki = kbase + j * 16 + lm;
        float val = s[j][r] * scale;
        s[j][r] = (ki <= qi) ? val : -1e30f;
      }

#pragma unroll
    for (int r = 0; r < 8; r++) {
      float mx = fmaxf(s[0][r], s[1][r]);
      for (int off = 1; off < 16; off <<= 1)
        mx = fmaxf(mx, __shfl_xor(mx, off, 32));
      float mnew = fmaxf(mrow[r], mx);
      float corr = __expf(mrow[r] - mnew);
      float p0 = __expf(s[0][r] - mnew);
      float p1 = __expf(s[1][r] - mnew);
      s[0][r] = p0; s[1][r] = p1;
      float rs = p0 + p1;
      for (int off = 1; off < 16; off <<= 1)
        rs += __shfl_xor(rs, off, 32);
      lrow[r] = lrow[r] * corr + rs;
      mrow[r] = mnew;
#pragma unroll
      for (int c2 = 0; c2 < 4; c2++) ctx[c2][r] *= corr;
    }

    // P: C-layout -> A-layout via per-wave LDS scratch
#pragma unroll
    for (int j = 0; j < 2; j++)
#pragma unroll
      for (int r = 0; r < 8; r++)
        Ps[wave][r + (hi << 3)][j * 16 + lm] = (half_t)s[j][r];
    Frag16 aP;
    aP.h[0] = *(const v8h*)&Ps[wave][lm][ka];
    aP.h[1] = *(const v8h*)&Ps[wave][lm][ka + 16];

#pragma unroll
    for (int c2 = 0; c2 < 4; c2++) {
      int drow = c2 * 16 + lm;
      Frag16 bV;
      bV.h[0] = *(const v8h*)&Vt[drow][kb];
      bV.h[1] = *(const v8h*)&Vt[drow][kb + 16];
      ctx[c2] = wmma_f16(aP.v, bV.v, ctx[c2]);
    }
  }

#pragma unroll
  for (int r = 0; r < 8; r++) {
    float inv = 1.0f / lrow[r];
    int m = q0 + r + (hi << 3);
#pragma unroll
    for (int c2 = 0; c2 < 4; c2++)
      O[headoff + (size_t)m * E_ + c2 * 16 + lm] = (half_t)(ctx[c2][r] * inv);
  }
}

// ---------------------------------------------------------------------------
// residual add + LayerNorm (one 1024-wide row per 256-thread block)
// ---------------------------------------------------------------------------
__global__ __launch_bounds__(256) void add_ln_kernel(
    const float* __restrict__ a, const float* __restrict__ res,
    const float* __restrict__ w, const float* __restrict__ b,
    float* __restrict__ outf, half_t* __restrict__ outh) {
  __shared__ float red[256];
  int row = blockIdx.x, tid = threadIdx.x;
  float vals[4];
  float s = 0.f;
#pragma unroll
  for (int i = 0; i < 4; i++) {
    int col = tid + i * 256;
    vals[i] = a[(size_t)row * E_ + col] + res[(size_t)row * E_ + col];
    s += vals[i];
  }
  red[tid] = s; __syncthreads();
  for (int off = 128; off > 0; off >>= 1) {
    if (tid < off) red[tid] += red[tid + off];
    __syncthreads();
  }
  float mu = red[0] * (1.f / E_);
  __syncthreads();
  float s2 = 0.f;
#pragma unroll
  for (int i = 0; i < 4; i++) { float d = vals[i] - mu; s2 += d * d; }
  red[tid] = s2; __syncthreads();
  for (int off = 128; off > 0; off >>= 1) {
    if (tid < off) red[tid] += red[tid + off];
    __syncthreads();
  }
  float rstd = rsqrtf(red[0] * (1.f / E_) + 1e-5f);
#pragma unroll
  for (int i = 0; i < 4; i++) {
    int col = tid + i * 256;
    float y = (vals[i] - mu) * rstd * w[col] + b[col];
    outf[(size_t)row * E_ + col] = y;
    if (outh) outh[(size_t)row * E_ + col] = (half_t)y;
  }
}

// ---------------------------------------------------------------------------
// gate logits -> softmax -> top-2 scatter into dense gsel[M,8]
// ---------------------------------------------------------------------------
__global__ __launch_bounds__(256) void gate_kernel(
    const float* __restrict__ x1, const float* __restrict__ gw,
    const float* __restrict__ gb, float* __restrict__ gsel) {
  __shared__ float red[256];
  __shared__ float logits[NE_];
  int row = blockIdx.x, tid = threadIdx.x;
  float xv[4];
#pragma unroll
  for (int i = 0; i < 4; i++) xv[i] = x1[(size_t)row * E_ + tid + i * 256];
  for (int e = 0; e < NE_; e++) {
    float p = 0.f;
#pragma unroll
    for (int i = 0; i < 4; i++) p += xv[i] * gw[(size_t)e * E_ + tid + i * 256];
    red[tid] = p; __syncthreads();
    for (int off = 128; off > 0; off >>= 1) {
      if (tid < off) red[tid] += red[tid + off];
      __syncthreads();
    }
    if (tid == 0) logits[e] = red[0] + gb[e];
    __syncthreads();
  }
  if (tid == 0) {
    float mx = logits[0];
    for (int e = 1; e < NE_; e++) mx = fmaxf(mx, logits[e]);
    float g[NE_]; float sum = 0.f;
    for (int e = 0; e < NE_; e++) { g[e] = __expf(logits[e] - mx); sum += g[e]; }
    float inv = 1.f / sum;
    for (int e = 0; e < NE_; e++) g[e] *= inv;
    int i0 = 0;
    for (int e = 1; e < NE_; e++) if (g[e] > g[i0]) i0 = e;
    int i1 = -1;
    for (int e = 0; e < NE_; e++) {
      if (e == i0) continue;
      if (i1 < 0 || g[e] > g[i1]) i1 = e;
    }
    for (int e = 0; e < NE_; e++)
      gsel[(size_t)row * NE_ + e] = (e == i0 || e == i1) ? g[e] : 0.f;
  }
}

// ---------------------------------------------------------------------------
extern "C" void kernel_launch(void* const* d_in, const int* in_sizes, int n_in,
                              void* d_out, int out_size, void* d_ws,
                              size_t ws_size, hipStream_t stream) {
  (void)in_sizes; (void)n_in; (void)out_size; (void)ws_size;
  const float* x          = (const float*)d_in[0];
  const float* in_proj_w  = (const float*)d_in[1];
  const float* in_proj_b  = (const float*)d_in[2];
  const float* out_proj_w = (const float*)d_in[3];
  const float* out_proj_b = (const float*)d_in[4];
  const float* gate_w     = (const float*)d_in[5];
  const float* gate_b     = (const float*)d_in[6];
  const float* expert_w   = (const float*)d_in[7];
  const float* expert_b   = (const float*)d_in[8];
  const float* ln1_w      = (const float*)d_in[9];
  const float* ln1_b      = (const float*)d_in[10];
  const float* ln2_w      = (const float*)d_in[11];
  const float* ln2_b      = (const float*)d_in[12];
  float* out = (float*)d_out;

  char* p = (char*)d_ws;
  auto take = [&](size_t bytes) {
    char* q = p;
    p += (bytes + 255) & ~(size_t)255;
    return q;
  };
  half_t* x_h    = (half_t*)take((size_t)MTOK * E_ * 2);
  half_t* qin_h  = (half_t*)take((size_t)MTOK * E_ * 2);
  half_t* q_h    = (half_t*)take((size_t)MTOK * E_ * 2);
  half_t* k_h    = (half_t*)take((size_t)MTOK * E_ * 2);
  half_t* v_h    = (half_t*)take((size_t)MTOK * E_ * 2);
  half_t* ctx_h  = (half_t*)take((size_t)MTOK * E_ * 2);
  float*  attn_o = (float*)take((size_t)MTOK * E_ * 4);
  float*  x1_f   = (float*)take((size_t)MTOK * E_ * 4);
  half_t* x1_h   = (half_t*)take((size_t)MTOK * E_ * 2);
  float*  gsel   = (float*)take((size_t)MTOK * NE_ * 4);
  float*  moe    = (float*)take((size_t)MTOK * E_ * 4);
  half_t* wqkv_h = (half_t*)take((size_t)3 * E_ * E_ * 2);
  half_t* wo_h   = (half_t*)take((size_t)E_ * E_ * 2);
  half_t* we_h   = (half_t*)take((size_t)NE_ * E_ * E_ * 2);

  // weight fp32 -> fp16 pre-conversion
  cast_f16_kernel<<<(3 * E_ * E_) / 1024, 256, 0, stream>>>(in_proj_w, wqkv_h);
  cast_f16_kernel<<<(E_ * E_) / 1024, 256, 0, stream>>>(out_proj_w, wo_h);
  cast_f16_kernel<<<(NE_ * E_ * E_) / 1024, 256, 0, stream>>>(expert_w, we_h);

  dim3 gemm_grid(MTOK / 128, E_ / 128);

  rope_cast_kernel<<<(MTOK * H_ * (D_ / 2)) / 256, 256, 0, stream>>>(x, x_h,
                                                                     qin_h);
  gemm_wmma_kernel<0, false><<<gemm_grid, 256, 0, stream>>>(
      qin_h, wqkv_h, in_proj_b, q_h, nullptr, nullptr, 0, E_, E_);
  gemm_wmma_kernel<0, false><<<gemm_grid, 256, 0, stream>>>(
      qin_h, wqkv_h + (size_t)E_ * E_, in_proj_b + E_, k_h, nullptr, nullptr, 0,
      E_, E_);
  gemm_wmma_kernel<0, false><<<gemm_grid, 256, 0, stream>>>(
      x_h, wqkv_h + (size_t)2 * E_ * E_, in_proj_b + 2 * E_, v_h, nullptr,
      nullptr, 0, E_, E_);
  attn_kernel<<<dim3(S_ / 64, B_ * H_), 128, 0, stream>>>(q_h, k_h, v_h, ctx_h);
  gemm_wmma_kernel<1, false><<<gemm_grid, 256, 0, stream>>>(
      ctx_h, wo_h, out_proj_b, nullptr, attn_o, nullptr, 0, E_, E_);
  add_ln_kernel<<<MTOK, 256, 0, stream>>>(x, attn_o, ln1_w, ln1_b, x1_f, x1_h);
  gate_kernel<<<MTOK, 256, 0, stream>>>(x1_f, gate_w, gate_b, gsel);
  gemm_wmma_kernel<2, true><<<gemm_grid, 256, 0, stream>>>(
      x1_h, we_h, expert_b, nullptr, moe, gsel, 0, E_, E_);
  for (int e = 1; e < NE_; e++)
    gemm_wmma_kernel<2, false><<<gemm_grid, 256, 0, stream>>>(
        x1_h, we_h + (size_t)e * E_ * E_, expert_b + (size_t)e * E_, nullptr,
        moe, gsel, e, E_, E_);
  add_ln_kernel<<<MTOK, 256, 0, stream>>>(x1_f, moe, ln2_w, ln2_b, out,
                                          nullptr);
}
